// Femto_Chatbot_32890859553476
// MI455X (gfx1250) — matrix operand order repaired and checked by reading, not compiled, not run on hardware
//
#include <hip/hip_runtime.h>
#include <hip/hip_bf16.h>

typedef __attribute__((ext_vector_type(16))) _Float16 v16h;
typedef __attribute__((ext_vector_type(8)))  _Float16 v8h;
typedef __attribute__((ext_vector_type(8)))  float    v8f;

#define VOC   50257
#define NPAD  50304   // 393 col-groups * 8 tiles * 16
#define DD    256
#define SEQ   4096
#define BATCH 2
#define MROWS (BATCH*SEQ)   // 8192
#define NCOLGROUPS 393      // ceil(ceil(50257/16)/8)

// workspace offsets in _Float16 units
#define Q_OFF   0L
#define K_OFF   2097152L
#define VT_OFF  4194304L
#define O_OFF   6291456L
#define WQT_OFF 8388608L
#define WKT_OFF 8454144L
#define WVT_OFF 8519680L
#define WPT_OFF 8585216L
// total = 8585216 + 50304*256 = 21463040 halves ~= 41 MiB

__device__ __forceinline__ v8f wmma16(v16h a, v16h b, v8f c) {
  return __builtin_amdgcn_wmma_f32_16x16x32_f16(false, a, false, b, (short)0, c,
                                                false, false);
}

// A-fragment (16x32 f16): lane row = lane%16; halves 0..15 = K kc+hi*16+j
// (contiguous k order, matching the B fragments)
__device__ __forceinline__ v16h load_a_frag(const _Float16* rowPtr, int kc, int hi) {
  union { v16h v; v8h h[2]; } u;
  u.h[0] = *(const v8h*)(rowPtr + kc + hi * 16);
  u.h[1] = *(const v8h*)(rowPtr + kc + hi * 16 + 8);
  return u.v;
}

// ---------------- prep: W[K=256][N] f32 -> WT[NP][256] f16 (zero-padded) -----
// One-shot streaming read of W: non-temporal loads keep L2 clean for later
// phases (WpT + E are the L2 working set that matters).
__global__ void transpose_w_kernel(const float* __restrict__ W,
                                   _Float16* __restrict__ WT,
                                   int N, long total) {
  long idx = (long)blockIdx.x * blockDim.x + threadIdx.x;
  if (idx >= total) return;
  int  k = (int)(idx & 255);
  long n = idx >> 8;
  float v = (n < (long)N) ? __builtin_nontemporal_load(W + (long)k * N + n) : 0.0f;
  WT[idx] = (_Float16)v;
}

// ---------------- fused embed + QKV GEMMs -----------------------------------
// one wave per 16-row tile; Q,K row-major f16; V stored transposed VT[b][d][t]
__global__ void qkv_kernel(const int* __restrict__ tokens,
                           const float* __restrict__ E,
                           const _Float16* __restrict__ WqT,
                           const _Float16* __restrict__ WkT,
                           const _Float16* __restrict__ WvT,
                           const float* __restrict__ bq,
                           const float* __restrict__ bk,
                           const float* __restrict__ bv,
                           _Float16* __restrict__ Q,
                           _Float16* __restrict__ K,
                           _Float16* __restrict__ VT) {
  int wave = (int)((blockIdx.x * blockDim.x + threadIdx.x) >> 5); // 0..511
  int lane = threadIdx.x & 31;
  int col  = lane & 15, hi = lane >> 4, kb = hi << 4;
  int r0   = wave * 16;

  // gather A fragments from embedding (f32 -> f16)
  int arow = r0 + (lane & 15);
  const float* erow = E + (long)tokens[arow] * DD;
  v16h a[8];
#pragma unroll
  for (int c = 0; c < 8; ++c) {
    union { v16h v; _Float16 h[16]; } u;
#pragma unroll
    for (int j = 0; j < 8; ++j) {
      u.h[j]     = (_Float16)erow[c * 32 + hi * 16 + j];
      u.h[j + 8] = (_Float16)erow[c * 32 + hi * 16 + 8 + j];
    }
    a[c] = u.v;
  }

#pragma unroll
  for (int m = 0; m < 3; ++m) {
    const _Float16* WT = (m == 0) ? WqT : (m == 1) ? WkT : WvT;
    const float*    bb = (m == 0) ? bq  : (m == 1) ? bk  : bv;
    for (int ct = 0; ct < 16; ++ct) {
      int n0 = ct * 16;
      v8f acc = {};
#pragma unroll
      for (int c = 0; c < 8; ++c) {
        v16h bf = *(const v16h*)(WT + (long)(n0 + col) * DD + c * 32 + kb);
        acc = wmma16(a[c], bf, acc);
      }
      float bias = bb[n0 + col];
      if (m == 2) { // V transposed store
#pragma unroll
        for (int v = 0; v < 8; ++v) {
          int row = r0 + v + hi * 8;
          int bt  = row >> 12;       // batch
          int t   = row & 4095;      // seq pos
          VT[((long)bt * DD + (n0 + col)) * SEQ + t] = (_Float16)(acc[v] + bias);
        }
      } else {
        _Float16* Out = (m == 0) ? Q : K;
#pragma unroll
        for (int v = 0; v < 8; ++v) {
          int row = r0 + v + hi * 8;
          Out[(long)row * DD + n0 + col] = (_Float16)(acc[v] + bias);
        }
      }
    }
  }
}

// ---------------- causal flash attention ------------------------------------
// one wave per 16-query tile; 32 keys per step; O f16 row-major
__global__ void attn_kernel(const _Float16* __restrict__ Q,
                            const _Float16* __restrict__ K,
                            const _Float16* __restrict__ VT,
                            _Float16* __restrict__ O) {
  __shared__ __align__(16) _Float16 lds[4][16 * 32];
  int wid  = threadIdx.x >> 5;
  int wave = (int)(blockIdx.x * (blockDim.x >> 5)) + wid; // 0..511
  int lane = threadIdx.x & 31;
  int col  = lane & 15, hi = lane >> 4, kb = hi << 4;
  int r0   = wave * 16;          // global row base
  int bt   = r0 >> 12;           // batch
  int qs   = r0 & 4095;          // seq position base
  const _Float16* Kb = K  + (long)bt * SEQ * DD;
  const _Float16* Vb = VT + (long)bt * DD * SEQ;
  _Float16* ldsw = lds[wid];

  const _Float16* qrow = Q + (long)(r0 + (lane & 15)) * DD;
  v16h qa[8];
#pragma unroll
  for (int c = 0; c < 8; ++c) qa[c] = load_a_frag(qrow, c * 32, hi);

  float mrow[8], lrow[8];
#pragma unroll
  for (int v = 0; v < 8; ++v) { mrow[v] = -__builtin_inff(); lrow[v] = 0.0f; }
  v8f acc[16];
  { v8f z = {};
#pragma unroll
    for (int d = 0; d < 16; ++d) acc[d] = z; }

  const float scale = 0.0625f;   // 1/sqrt(256)

  for (int kt = 0; kt < qs + 16; kt += 32) {
    v8f s0 = {}, s1 = {};
#pragma unroll
    for (int c = 0; c < 8; ++c) {
      v16h b0 = *(const v16h*)(Kb + (long)(kt + col) * DD + c * 32 + kb);
      s0 = wmma16(qa[c], b0, s0);
      v16h b1 = *(const v16h*)(Kb + (long)(kt + 16 + col) * DD + c * 32 + kb);
      s1 = wmma16(qa[c], b1, s1);
    }
#pragma unroll
    for (int v = 0; v < 8; ++v) {
      int qg = qs + v + hi * 8;                 // this row's seq position
      float x0 = s0[v] * scale, x1 = s1[v] * scale;
      if (kt + col      > qg) x0 = -__builtin_inff();
      if (kt + 16 + col > qg) x1 = -__builtin_inff();
      float t = fmaxf(x0, x1);
      t = fmaxf(t, __shfl_xor(t, 1, 16));
      t = fmaxf(t, __shfl_xor(t, 2, 16));
      t = fmaxf(t, __shfl_xor(t, 4, 16));
      t = fmaxf(t, __shfl_xor(t, 8, 16));
      float mnew  = fmaxf(mrow[v], t);
      float alpha = __expf(mrow[v] - mnew);
      float p0 = __expf(x0 - mnew);
      float p1 = __expf(x1 - mnew);
      float rs = p0 + p1;
      rs += __shfl_xor(rs, 1, 16);
      rs += __shfl_xor(rs, 2, 16);
      rs += __shfl_xor(rs, 4, 16);
      rs += __shfl_xor(rs, 8, 16);
      lrow[v] = lrow[v] * alpha + rs;
      mrow[v] = mnew;
      int rr = v + hi * 8;
      ldsw[rr * 32 + col]      = (_Float16)p0;
      ldsw[rr * 32 + 16 + col] = (_Float16)p1;
#pragma unroll
      for (int d = 0; d < 16; ++d) acc[d][v] *= alpha;
    }
    asm volatile("s_wait_dscnt 0" ::: "memory");   // cross-lane LDS visibility (per-wave, no barrier)
    v16h pa = load_a_frag(ldsw + (lane & 15) * 32, 0, hi);
#pragma unroll
    for (int d = 0; d < 16; ++d) {
      v16h vb = *(const v16h*)(Vb + (long)(d * 16 + col) * SEQ + kt + kb);
      acc[d] = wmma16(pa, vb, acc[d]);
    }
  }

#pragma unroll
  for (int v = 0; v < 8; ++v) {
    float inv = 1.0f / lrow[v];
    int row = r0 + v + hi * 8;
#pragma unroll
    for (int d = 0; d < 16; ++d)
      O[(long)row * DD + d * 16 + col] = (_Float16)(acc[d][v] * inv);
  }
}

// ---------------- logits = O @ Wp + bp --------------------------------------
// Store-bound: 1.65 GB f32 output streamed with non-temporal stores so the
// 26 MB WpT working set stays resident in the 192 MB L2 across row-tiles.
__global__ void logits_kernel(const _Float16* __restrict__ O,
                              const _Float16* __restrict__ WpT,
                              const float* __restrict__ bp,
                              float* __restrict__ out) {
  int gw   = (int)((blockIdx.x * blockDim.x + threadIdx.x) >> 5);
  int lane = threadIdx.x & 31;
  int col  = lane & 15, hi = lane >> 4, kb = hi << 4;
  int rowTile  = gw / NCOLGROUPS;
  int colGroup = gw % NCOLGROUPS;
  int r0 = rowTile * 16;

  const _Float16* orow = O + (long)(r0 + (lane & 15)) * DD;
  v16h a[8];
#pragma unroll
  for (int c = 0; c < 8; ++c) a[c] = load_a_frag(orow, c * 32, hi);

#pragma unroll
  for (int t = 0; t < 8; ++t) {
    int n0 = (colGroup * 8 + t) * 16;          // < NPAD by construction
    if (t < 7)
      __builtin_prefetch(WpT + (long)(n0 + 16 + col) * DD + kb, 0, 1);
    v8f acc = {};
#pragma unroll
    for (int c = 0; c < 8; ++c) {
      v16h bf = *(const v16h*)(WpT + (long)(n0 + col) * DD + c * 32 + kb);
      acc = wmma16(a[c], bf, acc);
    }
    int cg = n0 + col;
    if (cg < VOC) {                            // divergence only around stores
      float bias = bp[cg];
#pragma unroll
      for (int v = 0; v < 8; ++v) {
        int row = r0 + v + hi * 8;
        __builtin_nontemporal_store(acc[v] + bias, out + (long)row * VOC + cg);
      }
    }
  }
}

extern "C" void kernel_launch(void* const* d_in, const int* in_sizes, int n_in,
                              void* d_out, int out_size, void* d_ws, size_t ws_size,
                              hipStream_t stream) {
  const int*   tokens = (const int*)  d_in[0];
  const float* E      = (const float*)d_in[1];
  const float* Wq     = (const float*)d_in[2];
  const float* bq     = (const float*)d_in[3];
  const float* Wk     = (const float*)d_in[4];
  const float* bk     = (const float*)d_in[5];
  const float* Wv     = (const float*)d_in[6];
  const float* bv     = (const float*)d_in[7];
  const float* Wp     = (const float*)d_in[8];
  const float* bp     = (const float*)d_in[9];
  float* out = (float*)d_out;

  _Float16* w   = (_Float16*)d_ws;
  _Float16* Qb  = w + Q_OFF;
  _Float16* Kb  = w + K_OFF;
  _Float16* VTb = w + VT_OFF;
  _Float16* Ob  = w + O_OFF;
  _Float16* WqT = w + WQT_OFF;
  _Float16* WkT = w + WKT_OFF;
  _Float16* WvT = w + WVT_OFF;
  _Float16* WpT = w + WPT_OFF;

  // weight transpose/convert (f32 -> f16, Bt layout [N][K])
  transpose_w_kernel<<<256, 256, 0, stream>>>(Wq, WqT, DD, 65536L);
  transpose_w_kernel<<<256, 256, 0, stream>>>(Wk, WkT, DD, 65536L);
  transpose_w_kernel<<<256, 256, 0, stream>>>(Wv, WvT, DD, 65536L);
  transpose_w_kernel<<<NPAD, 256, 0, stream>>>(Wp, WpT, VOC, (long)NPAD * 256);

  // fused embed + QKV: 512 waves
  qkv_kernel<<<128, 128, 0, stream>>>(tokens, E, WqT, WkT, WvT, bq, bk, bv,
                                      Qb, Kb, VTb);
  // causal flash attention: 512 waves
  attn_kernel<<<128, 128, 0, stream>>>(Qb, Kb, VTb, Ob);
  // final projection: 512 row-tiles x 393 col-groups waves, 4 waves/block
  logits_kernel<<<(MROWS / 16) * NCOLGROUPS / 4, 128, 0, stream>>>(Ob, WpT, bp, out);
}
